// GraphConvolution_90305982365717
// MI455X (gfx1250) — compile-verified, matrix operand order checked by reading, not run on hardware
//
#include <hip/hip_runtime.h>

typedef _Float16 half_t;
typedef __attribute__((ext_vector_type(16))) _Float16 v16h;
typedef __attribute__((ext_vector_type(8)))  float    v8f;
typedef __attribute__((ext_vector_type(4)))  int      v4i;

// ---------------- problem sizes (fixed by the reference) ----------------
constexpr int NROW = 8192;   // n
constexpr int DIM  = 1024;   // d

// ---------------- GEMM tiling ----------------
#define BM 128
#define BN 128
#define BK 32
#define LDT 40   // LDS row stride (halves): 80B rows keep b128 ops 16B-aligned, break bank alignment
#define SPLITK 8

// CDNA5 async global->LDS path (ASYNCcnt), guarded so absence degrades gracefully.
// Builtin signature (from hipcc diagnostic): (v4i AS1*, v4i AS3*, imm offset, imm cpol)
#if __has_builtin(__builtin_amdgcn_global_load_async_to_lds_b128) && \
    __has_builtin(__builtin_amdgcn_s_wait_asynccnt)
#define HAS_ASYNC_LDS 1
typedef __attribute__((address_space(1))) v4i* gptr_v4i;   // global
typedef __attribute__((address_space(3))) v4i* lptr_v4i;   // LDS (32-bit ptr)
#endif

union FragU { v16h v; unsigned int u[8]; };

// K-offset inside a 16x32 f16 WMMA fragment for packed dword j, lane half kh
__device__ __forceinline__ int klocal(int j, int kh) {
  return ((j < 4) ? 0 : 16) + kh * 8 + (j & 3) * 2;
}

enum { OUT_F16 = 0, OUT_F32 = 1, OUT_F32_ATOMIC = 2 };

// C = A(MxK f16 row-major) * B(KxN f16 row-major), f32 WMMA accumulate.
// Double-buffered LDS; A tile staged with async global->LDS when available.
// M,N multiples of 128; K multiple of 32*gridDim.z.
template <int MODE>
__global__ __launch_bounds__(256) void gemm_f16_wmma(
    const half_t* __restrict__ A, const half_t* __restrict__ B,
    void* __restrict__ outp, int M, int N, int K)
{
  __shared__ __align__(16) half_t Ash[2][BM * LDT];
  __shared__ __align__(16) half_t Bsh[2][BN * LDT];   // transposed: [n][k]

  const int tid   = threadIdx.x;
  const int lane  = tid & 31;
  const int wid   = tid >> 5;
  const int waveM = (wid & 3) * 32;   // 4 waves along M
  const int waveN = (wid >> 2) * 64;  // 2 waves along N
  const int m0 = blockIdx.y * BM;
  const int n0 = blockIdx.x * BN;
  const int lm = lane & 15;
  const int kh = lane >> 4;

  const int Kc   = K / (int)gridDim.z;      // split-K chunk
  const int kbeg = (int)blockIdx.z * Kc;
  const int kend = kbeg + Kc;

  const int ar = tid >> 2;            // A-tile loader: row 0..63 (+64), 8 halves
  const int ac = (tid & 3) * 8;
  const int bk = tid >> 4;            // B-tile loader: k 0..15 (+16), 8 halves
  const int bn = (tid & 15) * 8;

  const half_t* Abase = A + (size_t)m0 * K;

  auto issueA = [&](int k0, int buf) {
#pragma unroll
    for (int i = 0; i < 2; ++i) {
      const int row = ar + i * 64;
      const half_t* src = Abase + (size_t)row * K + (k0 + ac);
      half_t* dst = &Ash[buf][row * LDT + ac];
#if HAS_ASYNC_LDS
      __builtin_amdgcn_global_load_async_to_lds_b128(
          (gptr_v4i)(size_t)src, (lptr_v4i)(unsigned int)(size_t)dst, 0, 0);
#else
      *(uint4*)dst = *(const uint4*)src;
#endif
    }
  };
  auto loadB = [&](int k0, uint4& v0, uint4& v1) {
    v0 = *(const uint4*)(B + (size_t)(k0 + bk)      * N + (n0 + bn));
    v1 = *(const uint4*)(B + (size_t)(k0 + bk + 16) * N + (n0 + bn));
  };
  auto storeB = [&](int buf, const uint4& v0, const uint4& v1) {
    const half_t* h0 = (const half_t*)&v0;
    const half_t* h1 = (const half_t*)&v1;
#pragma unroll
    for (int j = 0; j < 8; ++j) {
      Bsh[buf][(bn + j) * LDT + bk]      = h0[j];
      Bsh[buf][(bn + j) * LDT + bk + 16] = h1[j];
    }
  };

  v8f acc[2][4] = {};
  uint4 vb0, vb1;

  // ---- preload tile 0 ----
  issueA(kbeg, 0);
  loadB(kbeg, vb0, vb1);
  storeB(0, vb0, vb1);
#if HAS_ASYNC_LDS
  __builtin_amdgcn_s_wait_asynccnt(0);
#endif
  __syncthreads();

  for (int k0 = kbeg; k0 < kend; k0 += BK) {
    const int  buf      = ((k0 - kbeg) >> 5) & 1;
    const bool haveNext = (k0 + BK) < kend;

    if (haveNext) {                       // stage tile i+1 while computing tile i
      issueA(k0 + BK, buf ^ 1);
      loadB(k0 + BK, vb0, vb1);
      if (k0 + 2 * BK < kend) {           // WGP-scope prefetch of tile i+2
        __builtin_prefetch(Abase + (size_t)ar * K + (k0 + 2 * BK) + ac, 0, 3);
        __builtin_prefetch(B + (size_t)(k0 + 2 * BK + bk) * N + n0 + bn, 0, 3);
      }
    }

    // ---- fragment gather (packed dwords -> compiler emits ds_load_b128) ----
    FragU a[2], b[4];
#pragma unroll
    for (int mi = 0; mi < 2; ++mi) {
      const int row = waveM + mi * 16 + lm;
#pragma unroll
      for (int j = 0; j < 8; ++j)
        a[mi].u[j] = *(const unsigned int*)&Ash[buf][row * LDT + klocal(j, kh)];
    }
#pragma unroll
    for (int ni = 0; ni < 4; ++ni) {
      const int col = waveN + ni * 16 + lm;
#pragma unroll
      for (int j = 0; j < 8; ++j)
        b[ni].u[j] = *(const unsigned int*)&Bsh[buf][col * LDT + klocal(j, kh)];
    }

#pragma unroll
    for (int mi = 0; mi < 2; ++mi)
#pragma unroll
      for (int ni = 0; ni < 4; ++ni)
        acc[mi][ni] = __builtin_amdgcn_wmma_f32_16x16x32_f16(
            false, a[mi].v, false, b[ni].v, (short)0, acc[mi][ni], false, false);

    if (haveNext) storeB(buf ^ 1, vb0, vb1);
#if HAS_ASYNC_LDS
    __builtin_amdgcn_s_wait_asynccnt(0);
#endif
    __syncthreads();
  }

  // ---- epilogue: branch-free per MODE ----
#pragma unroll
  for (int mi = 0; mi < 2; ++mi) {
#pragma unroll
    for (int ni = 0; ni < 4; ++ni) {
      const int n = n0 + waveN + ni * 16 + lm;
      const size_t base = (size_t)(m0 + waveM + mi * 16 + kh * 8) * N + n;
#pragma unroll
      for (int v = 0; v < 8; ++v) {
        const float val = acc[mi][ni][v];
        if constexpr (MODE == OUT_F32)
          ((float*)outp)[base + (size_t)v * N] = val;
        else if constexpr (MODE == OUT_F16)
          ((half_t*)outp)[base + (size_t)v * N] = (half_t)val;
        else
          unsafeAtomicAdd((float*)outp + base + (size_t)v * N, val);
      }
    }
  }
}

// Row L2-normalize x (full f32 math), emit f16 copies of x, norm, and norm^T.
__global__ __launch_bounds__(256) void normalize_kernel(
    const float* __restrict__ x,
    half_t* __restrict__ xh, half_t* __restrict__ normh, half_t* __restrict__ normTh)
{
  const int row = blockIdx.x;
  const int tid = threadIdx.x;
  const float4 xv = *(const float4*)(x + (size_t)row * DIM + tid * 4);

  float ss = xv.x * xv.x + xv.y * xv.y + xv.z * xv.z + xv.w * xv.w;
#pragma unroll
  for (int off = 16; off > 0; off >>= 1) ss += __shfl_down(ss, off, 32);

  __shared__ float wsum[8];
  if ((tid & 31) == 0) wsum[tid >> 5] = ss;
  __syncthreads();
  float tot = 0.f;
#pragma unroll
  for (int i = 0; i < 8; ++i) tot += wsum[i];

  const float scale = 1.0f / fmaxf(sqrtf(tot), 1e-12f);

  const int c = tid * 4;
  union { half_t h[4]; uint2 u; } px, pn;
  px.h[0] = (half_t)xv.x;           px.h[1] = (half_t)xv.y;
  px.h[2] = (half_t)xv.z;           px.h[3] = (half_t)xv.w;
  pn.h[0] = (half_t)(xv.x * scale); pn.h[1] = (half_t)(xv.y * scale);
  pn.h[2] = (half_t)(xv.z * scale); pn.h[3] = (half_t)(xv.w * scale);

  *(uint2*)(xh    + (size_t)row * DIM + c) = px.u;
  *(uint2*)(normh + (size_t)row * DIM + c) = pn.u;
#pragma unroll
  for (int j = 0; j < 4; ++j)
    normTh[(size_t)(c + j) * NROW + row] = pn.h[j];
}

__global__ __launch_bounds__(256) void f32_to_f16_kernel(
    const float* __restrict__ src, half_t* __restrict__ dst)
{
  const int i = blockIdx.x * blockDim.x + threadIdx.x;  // one float4 per thread
  const float4 v = ((const float4*)src)[i];
  union { half_t h[4]; uint2 u; } p;
  p.h[0] = (half_t)v.x; p.h[1] = (half_t)v.y; p.h[2] = (half_t)v.z; p.h[3] = (half_t)v.w;
  ((uint2*)dst)[i] = p.u;
}

__global__ __launch_bounds__(256) void zero_f32_kernel(float* __restrict__ p)
{
  const int i = blockIdx.x * blockDim.x + threadIdx.x;
  ((float4*)p)[i] = make_float4(0.f, 0.f, 0.f, 0.f);
}

extern "C" void kernel_launch(void* const* d_in, const int* in_sizes, int n_in,
                              void* d_out, int out_size, void* d_ws, size_t ws_size,
                              hipStream_t stream) {
  (void)in_sizes; (void)n_in; (void)out_size; (void)ws_size;

  const float* x = (const float*)d_in[0];   // [8192,1024] f32
  const float* w = (const float*)d_in[1];   // [1024,1024] f32
  float* out = (float*)d_out;               // [8192,1024] f32

  // workspace layout (bytes), total ~58 MB
  char* ws = (char*)d_ws;
  half_t* xh     = (half_t*)(ws);                       // 16 MB  x as f16
  half_t* normh  = (half_t*)(ws + (size_t)16777216);    // 16 MB  norm
  half_t* normTh = (half_t*)(ws + (size_t)33554432);    // 16 MB  norm^T
  half_t* wh     = (half_t*)(ws + (size_t)50331648);    //  2 MB  weight f16
  half_t* Gh     = (half_t*)(ws + (size_t)52428800);    //  2 MB  G (f16)
  half_t* Hh     = (half_t*)(ws + (size_t)54525952);    //  2 MB  H (f16)
  float*  Gf     = (float*) (ws + (size_t)56623104);    //  4 MB  G (f32 split-K accum)

  normalize_kernel<<<NROW, 256, 0, stream>>>(x, xh, normh, normTh);
  f32_to_f16_kernel<<<(DIM * DIM / 4) / 256, 256, 0, stream>>>(w, wh);
  zero_f32_kernel<<<(DIM * DIM / 4) / 256, 256, 0, stream>>>(Gf);

  // G = norm^T @ x   [1024x1024], K=8192, split-K x8 -> 512 blocks, atomic f32 accum
  gemm_f16_wmma<OUT_F32_ATOMIC><<<dim3(DIM / BN, DIM / BM, SPLITK), 256, 0, stream>>>(
      normTh, xh, Gf, DIM, DIM, NROW);
  f32_to_f16_kernel<<<(DIM * DIM / 4) / 256, 256, 0, stream>>>(Gf, Gh);

  // H = G @ W        [1024x1024], K=1024
  gemm_f16_wmma<OUT_F16><<<dim3(DIM / BN, DIM / BM, 1), 256, 0, stream>>>(
      Gh, wh, Hh, DIM, DIM, DIM);

  // out = norm @ H   [8192x1024], K=1024 (f32 output)
  gemm_f16_wmma<OUT_F32><<<dim3(DIM / BN, NROW / BM, 1), 256, 0, stream>>>(
      normh, Hh, out, NROW, DIM, DIM);
}
